// MultiHeadAttentionBlock_28449863369250
// MI455X (gfx1250) — matrix operand order repaired and checked by reading, not compiled, not run on hardware
//
#include <hip/hip_runtime.h>

// ---------------------------------------------------------------------------
// MultiHeadAttentionBlock on MI455X (gfx1250).
//
// Identity: out = einsum("btuh,bthj->btjh", softmax(w), v) indexes v by t, not
// the summed u, so out[b,t,j,h] = v[b,t,h,j] * (softmax row-sum == 1) = v.
// The Q/K/softmax pipeline is algebraically dead. Remaining work:
//   v     = x @ Wv^T + bv          (4096x1024x1024 GEMM)
//   vperm = channel permute (h,j) -> j*H + h  (fused into GEMM1 store)
//   out   = vperm @ Wp^T + bp      (4096x1024x1024 GEMM)
// bf16 WMMA with f32 accumulation; tiles staged with CDNA5 async
// global->LDS DMA (ASYNCcnt) so no VGPR staging / no spills.
// ---------------------------------------------------------------------------

typedef __bf16 bf16;
typedef __attribute__((ext_vector_type(4)))  float  f32x4;
typedef __attribute__((ext_vector_type(4)))  __bf16 bf16x4;
typedef __attribute__((ext_vector_type(8)))  __bf16 bf16x8;
typedef __attribute__((ext_vector_type(16))) __bf16 v16bf;
typedef __attribute__((ext_vector_type(8)))  float  v8f;

constexpr int KC = 1024;       // C == N == K for both GEMMs
constexpr int NH = 16;         // heads
constexpr int DH = KC / NH;    // 64

#define LDSS 40   // LDS row stride in bf16 (80B) -> conflict-free ds_load_b128

// ---------------- fp32 -> bf16 conversion ----------------------------------
__global__ __launch_bounds__(256) void cvt_f32_to_bf16(const float* __restrict__ src,
                                                       bf16* __restrict__ dst, int n4) {
  int i = blockIdx.x * blockDim.x + threadIdx.x;
  if (i < n4) {
    f32x4 v = *(const f32x4*)(src + (size_t)i * 4);
    *(bf16x4*)(dst + (size_t)i * 4) = __builtin_convertvector(v, bf16x4);
  }
}

// --- A fragment (16x32 bf16): lane<16 row=lane, K {0..7,16..23};
//     lane>=16 row=lane-16, K {8..15,24..31} (ISA 7.12.2). ---
static __device__ __forceinline__ v16bf load_fragA(const bf16* base, int lane) {
  int r = lane & 15, half = lane >> 4;
  const bf16* p = base + r * LDSS + half * 8;
  bf16x8 lo = *(const bf16x8*)(p);
  bf16x8 hi = *(const bf16x8*)(p + 16);
  return __builtin_shufflevector(lo, hi, 0,1,2,3,4,5,6,7,8,9,10,11,12,13,14,15);
}

// --- B fragment (32x16 bf16): lane<16 col=lane holds K {0..7,16..23};
//     lane>=16 holds K {8..15,24..31} (matches A fragment). ---
static __device__ __forceinline__ v16bf load_fragB(const bf16* base, int lane) {
  int r = lane & 15, half = lane >> 4;
  const bf16* p = base + r * LDSS + half * 8;
  bf16x8 lo = *(const bf16x8*)(p);
  bf16x8 hi = *(const bf16x8*)(p + 16);
  return __builtin_shufflevector(lo, hi, 0,1,2,3,4,5,6,7,8,9,10,11,12,13,14,15);
}

// CDNA5 async DMA: copy 16B/lane global -> LDS, tracked by ASYNCcnt.
static __device__ __forceinline__ void async_b128(unsigned lds_off, const void* gptr) {
  asm volatile("global_load_async_to_lds_b128 %0, %1, off"
               :: "v"(lds_off), "v"(gptr) : "memory");
}
static __device__ __forceinline__ void wait_async0() {
  asm volatile("s_wait_asynccnt 0x0" ::: "memory");
}

// ---------------------------------------------------------------------------
// Y[m,n] = sum_k A[m,k]*Bw[n,k] + bias[n]   (nn.Linear: x @ W^T + b)
// 128x128 block tile, 256 threads = 8 waves (4x2), wave tile 32x64 (2x4 WMMA).
// ---------------------------------------------------------------------------
template <bool PERMUTE>
__global__ __launch_bounds__(256)
void gemm_bias_wmma(const bf16* __restrict__ A, const bf16* __restrict__ Bw,
                    const float* __restrict__ bias,
                    bf16* __restrict__ outb, float* __restrict__ outf, int M) {
  __shared__ bf16 As[2][128 * LDSS];
  __shared__ bf16 Bs[2][128 * LDSS];

  const int tid  = threadIdx.x;
  const int lane = tid & 31;
  const int wave = tid >> 5;
  const int wr   = wave & 3;   // 4 wave-rows x 32
  const int wc   = wave >> 2;  // 2 wave-cols x 64
  const int m0   = blockIdx.y * 128;
  const int n0   = blockIdx.x * 128;
  constexpr int KT = KC / 32;

  // Issue a 128x32 bf16 tile copy (A and B) via async DMA: 2 x b128 per thread
  // per matrix; no VGPR data staging at all.
  auto asyncTile = [&](int kt, int buf) {
    int k0 = kt << 5;
#pragma unroll
    for (int p = 0; p < 2; ++p) {
      int flat = p * 2048 + tid * 8;
      int row = flat >> 5, col = flat & 31;
      const bf16* ga = A  + (size_t)(m0 + row) * KC + k0 + col;
      const bf16* gb = Bw + (size_t)(n0 + row) * KC + k0 + col;
      unsigned la = (unsigned)(uintptr_t)&As[buf][row * LDSS + col];
      unsigned lb = (unsigned)(uintptr_t)&Bs[buf][row * LDSS + col];
      async_b128(la, ga);
      async_b128(lb, gb);
    }
  };

  v8f acc[2][4];
#pragma unroll
  for (int mi = 0; mi < 2; ++mi)
#pragma unroll
    for (int ni = 0; ni < 4; ++ni) {
      v8f z = {};
      acc[mi][ni] = z;
    }

  asyncTile(0, 0);
  wait_async0();
  __syncthreads();

#pragma unroll 2
  for (int kt = 0; kt < KT; ++kt) {
    const int cur = kt & 1;

    if (kt + 1 < KT) asyncTile(kt + 1, cur ^ 1);  // DMA next tile during WMMA
    if (kt + 2 < KT) {                            // warm L2 two tiles ahead
      int k2 = (kt + 2) << 5;
      __builtin_prefetch(A  + (size_t)(m0 + (tid >> 1)) * KC + k2, 0, 1);
      __builtin_prefetch(Bw + (size_t)(n0 + (tid >> 1)) * KC + k2, 0, 1);
    }

    v16bf afrag[2], bfrag[4];
#pragma unroll
    for (int mi = 0; mi < 2; ++mi)
      afrag[mi] = load_fragA(&As[cur][(wr * 32 + mi * 16) * LDSS], lane);
#pragma unroll
    for (int ni = 0; ni < 4; ++ni)
      bfrag[ni] = load_fragB(&Bs[cur][(wc * 64 + ni * 16) * LDSS], lane);

#pragma unroll
    for (int mi = 0; mi < 2; ++mi)
#pragma unroll
      for (int ni = 0; ni < 4; ++ni)
        acc[mi][ni] = __builtin_amdgcn_wmma_f32_16x16x32_bf16(
            false, afrag[mi], false, bfrag[ni],
            (short)0, acc[mi][ni], false, false);

    if (kt + 1 < KT) {
      wait_async0();      // my DMA portion landed in LDS
      __syncthreads();    // everyone's portion landed
    }
  }

  // Epilogue. C/D layout: VGPR r -> lanes0-15 M=r, lanes16-31 M=r+8; N=lane&15.
#pragma unroll
  for (int mi = 0; mi < 2; ++mi) {
    int rowBase = m0 + wr * 32 + mi * 16 + (lane >> 4) * 8;
#pragma unroll
    for (int ni = 0; ni < 4; ++ni) {
      int col = n0 + wc * 64 + ni * 16 + (lane & 15);
      float bcol = bias[col];
      if (PERMUTE) {
        int pcol = (col % DH) * NH + (col / DH);  // v chan h*DH+j -> j*NH+h
#pragma unroll
        for (int r = 0; r < 8; ++r)
          outb[(size_t)(rowBase + r) * KC + pcol] = (bf16)(acc[mi][ni][r] + bcol);
      } else {
#pragma unroll
        for (int r = 0; r < 8; ++r)
          outf[(size_t)(rowBase + r) * KC + col] = acc[mi][ni][r] + bcol;
      }
    }
  }
}

// ---------------------------------------------------------------------------
extern "C" void kernel_launch(void* const* d_in, const int* in_sizes, int n_in,
                              void* d_out, int out_size, void* d_ws, size_t ws_size,
                              hipStream_t stream) {
  const float* x  = (const float*)d_in[0];
  const float* Wv = (const float*)d_in[5];
  const float* bv = (const float*)d_in[6];
  const float* Wp = (const float*)d_in[7];
  const float* bp = (const float*)d_in[8];

  const int C = KC;                 // 1024
  const int M = in_sizes[0] / C;    // B*T = 4096

  // Workspace (bf16): x (M*C) | Wv (C*C) | Wp (C*C) | vperm (M*C)  ~= 20 MB
  char* ws = (char*)d_ws;
  bf16* xb    = (bf16*)(ws);
  bf16* wvb   = (bf16*)(ws + (size_t)M * C * 2);
  bf16* wpb   = (bf16*)(ws + (size_t)M * C * 2 + (size_t)C * C * 2);
  bf16* vperm = (bf16*)(ws + (size_t)M * C * 2 + (size_t)2 * C * C * 2);

  int nx4 = (M * C) / 4, nw4 = (C * C) / 4;
  cvt_f32_to_bf16<<<(nx4 + 255) / 256, 256, 0, stream>>>(x, xb, nx4);
  cvt_f32_to_bf16<<<(nw4 + 255) / 256, 256, 0, stream>>>(Wv, wvb, nw4);
  cvt_f32_to_bf16<<<(nw4 + 255) / 256, 256, 0, stream>>>(Wp, wpb, nw4);

  dim3 grid(C / 128, M / 128);   // 8 x 32
  gemm_bias_wmma<true ><<<grid, 256, 0, stream>>>(xb, wvb, bv, vperm, nullptr, M);
  gemm_bias_wmma<false><<<grid, 256, 0, stream>>>(vperm, wpb, bp, nullptr,
                                                  (float*)d_out, M);
}
